// Model_45870250721822
// MI455X (gfx1250) — compile-verified, matrix operand order checked by reading, not run on hardware
//
#include <hip/hip_runtime.h>
#include <hip/hip_bf16.h>
#include <math.h>

// ---------------------------------------------------------------------------
// TimesNet forward for MI455X (gfx1250, wave32, WMMA).
//   - inception(6 kernels) folded into one effective 11x11 conv.
//   - conv inputs staged into zero-padded [B][3072][64] f16 layout so the
//     121-offset implicit-GEMM loop is pure pointer-increment + WMMA
//     (v_wmma_f32_16x16x32_f16, f32 accum, EXEC always full, no cndmask).
//   - head GEMMs (bandwidth bound, M=16=batch) with v_wmma_f32_16x16x4_f32
//     on fp32 weights directly (no conversion traffic), split-K=8 + atomics.
//   - data-dependent top-k periods resolved on-device; every grid is static
//     (worst case) so the whole launch sequence is graph-capturable.
// ---------------------------------------------------------------------------

typedef __attribute__((ext_vector_type(16))) _Float16 v16h;
typedef __attribute__((ext_vector_type(8)))  float    v8f;
typedef __attribute__((ext_vector_type(2)))  float    v2f;

#define SEQ 256
#define DM  64
#define NB  16          // batch
#define MID 10125       // int(64*256*0.618)
#define MIDP 10128      // padded to /16
#define NTILE 633       // ceil(10125/16)
#define GEMM_SPLITS 8
#define PADP 3072       // padded positions per batch: max (H+10)*(W+10) = 2926

union F16x16 { v16h v; int4 q[2]; };
union H8     { _Float16 h[8]; int4 q; };

__device__ __forceinline__ float gelu_exact(float x) {
    return 0.5f * x * (1.0f + erff(x * 0.70710678118654752f));
}

// ---------------------------------------------------------------------------
// Token embedding (circular conv1d k=3, no bias) + positional encoding.
// grid: 16*256 blocks x 64 threads
// ---------------------------------------------------------------------------
__global__ void token_pe_kernel(const float* __restrict__ x,   // [16,256,16]
                                const float* __restrict__ tw,  // [64,16,3]
                                float* __restrict__ enc) {     // [16,256,64]
    int bt = blockIdx.x;
    int b = bt >> 8, t = bt & 255;
    int d = threadIdx.x;
    float s = 0.0f;
    #pragma unroll
    for (int j = 0; j < 3; ++j) {
        int tt = (t + 255 + j) & 255;   // circular pad=1
        const float* xp = x + ((size_t)b * SEQ + tt) * 16;
        const float* wp = tw + d * 48 + j;
        #pragma unroll
        for (int c = 0; c < 16; ++c) s += xp[c] * wp[c * 3];
    }
    int i = d >> 1;
    float freq = expf(-(float)(2 * i) * (9.210340371976184f / 64.0f)); // ln(1e4)
    float ang = (float)t * freq;
    float pe = (d & 1) ? cosf(ang) : sinf(ang);
    enc[((size_t)b * SEQ + t) * DM + d] = s + pe;
}

// ---------------------------------------------------------------------------
// Direct 256-pt rDFT amplitude: amp[b][f] = mean_n |X[b,f,n]|, f in [0,128].
// grid: (16,129) blocks x 64 threads
// ---------------------------------------------------------------------------
__global__ void dft_amp_kernel(const float* __restrict__ enc, float* __restrict__ amp) {
    int b = blockIdx.x, f = blockIdx.y, n = threadIdx.x;
    float w = -6.283185307179586f * (float)f / 256.0f;
    float re = 0.0f, im = 0.0f;
    for (int t = 0; t < SEQ; ++t) {
        float v = enc[((size_t)b * SEQ + t) * DM + n];
        float s, c;
        sincosf(w * (float)t, &s, &c);
        re += v * c;
        im += v * s;
    }
    __shared__ float sh[64];
    sh[n] = sqrtf(re * re + im * im);
    __syncthreads();
    for (int st = 32; st > 0; st >>= 1) {
        if (n < st) sh[n] += sh[n + st];
        __syncthreads();
    }
    if (n == 0) amp[b * 129 + f] = sh[0] * (1.0f / 64.0f);
}

// ---------------------------------------------------------------------------
// fl = mean_b amp, fl[0]=0, top-3 argmax -> bp[j]={f,period,H,L}; softmax sw.
// one block of 160 threads
// ---------------------------------------------------------------------------
__global__ void topk_kernel(const float* __restrict__ amp, int* __restrict__ bp,
                            float* __restrict__ sw) {
    __shared__ float fl[129];
    __shared__ int top[3];
    int t = threadIdx.x;
    if (t < 129) {
        float s = 0.0f;
        for (int b = 0; b < NB; ++b) s += amp[b * 129 + t];
        fl[t] = (t == 0) ? 0.0f : s * (1.0f / 16.0f);
    }
    __syncthreads();
    if (t == 0) {
        for (int j = 0; j < 3; ++j) {
            float best = -1e30f; int bi = 1;
            for (int f = 0; f < 129; ++f)
                if (fl[f] > best) { best = fl[f]; bi = f; }
            fl[bi] = -1e30f;
            if (bi < 1) bi = 1;
            top[j] = bi;
            int p = SEQ / bi;
            int H = (SEQ + p - 1) / p;
            bp[j * 4 + 0] = bi;
            bp[j * 4 + 1] = p;
            bp[j * 4 + 2] = H;
            bp[j * 4 + 3] = H * p;   // L
        }
    }
    __syncthreads();
    if (t < NB) {
        float v0 = amp[t * 129 + top[0]];
        float v1 = amp[t * 129 + top[1]];
        float v2 = amp[t * 129 + top[2]];
        float m = fmaxf(v0, fmaxf(v1, v2));
        float e0 = expf(v0 - m), e1 = expf(v1 - m), e2 = expf(v2 - m);
        float inv = 1.0f / (e0 + e1 + e2);
        sw[t * 3 + 0] = e0 * inv;
        sw[t * 3 + 1] = e1 * inv;
        sw[t * 3 + 2] = e2 * inv;
    }
}

// ---------------------------------------------------------------------------
// Fold 6 inception kernels (k=1..11, centered) into one 11x11 effective
// kernel (f16), layout [off=dy*11+dx][cout][cin]; beff = mean bias.
// ---------------------------------------------------------------------------
__device__ __forceinline__ float pick_w(const float* __restrict__ w, int k,
                                        int co, int ci, int dy, int dx) {
    int o = (11 - k) >> 1;
    int r = dy - o, c = dx - o;
    return (r >= 0 && r < k && c >= 0 && c < k)
               ? w[((size_t)(co * 64 + ci) * k + r) * k + c] : 0.0f;
}

__global__ void make_weff_kernel(const float* w0, const float* w1, const float* w2,
                                 const float* w3, const float* w4, const float* w5,
                                 const float* b0, const float* b1, const float* b2,
                                 const float* b3, const float* b4, const float* b5,
                                 _Float16* __restrict__ weff, float* __restrict__ beff) {
    int i = blockIdx.x * blockDim.x + threadIdx.x;
    if (i < 121 * 64 * 64) {
        int off = i >> 12, rem = i & 4095;
        int co = rem >> 6, ci = rem & 63;
        int dy = off / 11, dx = off - (off / 11) * 11;
        float s = pick_w(w0, 1, co, ci, dy, dx) + pick_w(w1, 3, co, ci, dy, dx) +
                  pick_w(w2, 5, co, ci, dy, dx) + pick_w(w3, 7, co, ci, dy, dx) +
                  pick_w(w4, 9, co, ci, dy, dx) + pick_w(w5, 11, co, ci, dy, dx);
        weff[i] = (_Float16)(s * (1.0f / 6.0f));
    }
    if (i < 64)
        beff[i] = (b0[i] + b1[i] + b2[i] + b3[i] + b4[i] + b5[i]) * (1.0f / 6.0f);
}

// ---------------------------------------------------------------------------
// acc = enc (residual base).  grid: 262144/256
// ---------------------------------------------------------------------------
__global__ void copy_acc_kernel(const float* __restrict__ enc, float* __restrict__ acc) {
    int i = blockIdx.x * blockDim.x + threadIdx.x;
    acc[i] = enc[i];
}

// ---------------------------------------------------------------------------
// Stage enc into zero-padded per-branch layout:
// encpad[b][pp][n], pp = (hh+5)*(W+10) + (ww+5); zero in halo and for t>=256
// (this realizes both jnp.pad time-padding and conv2d "same" zero padding).
// grid: (16, 3072) x 64
// ---------------------------------------------------------------------------
__global__ void stage_encpad_kernel(const float* __restrict__ enc,
                                    const int* __restrict__ bp, int br,
                                    _Float16* __restrict__ encpad) {
    int b = blockIdx.x, pp = blockIdx.y, n = threadIdx.x;
    int period = bp[br * 4 + 1], H = bp[br * 4 + 2];
    int Wp = period + 10;
    int hh = pp / Wp - 5;
    int ww = pp - (pp / Wp) * Wp - 5;
    int t = hh * period + ww;
    bool v = (hh >= 0) && (hh < H) && (ww >= 0) && (ww < period) && (t < SEQ);
    float x = v ? enc[((size_t)b * SEQ + t) * DM + n] : 0.0f;
    encpad[((size_t)b * PADP + pp) * DM + n] = (_Float16)x;
}

// zero a buffer of int4 chunks
__global__ void zero_b16_kernel(int4* __restrict__ p, int n4) {
    int i = blockIdx.x * blockDim.x + threadIdx.x;
    if (i < n4) p[i] = make_int4(0, 0, 0, 0);
}

// ---------------------------------------------------------------------------
// Implicit-GEMM conv tile on padded input: D[16 cout x 16 positions], 121
// offsets x 2 K-chunks, pure pointer-bump addressing -> WMMA dominates.
// A (16x32 f16): lane<16 row M=lane, K {0..7,16..23}; lane>=16 same row,
//                K {8..15,24..31}.  B mirrors with lane = position column.
// ---------------------------------------------------------------------------
__device__ __forceinline__ v8f conv_tile(const _Float16* __restrict__ srcpad,
                                         const _Float16* __restrict__ weff,
                                         int b, int ct, int pt, int period, int lane) {
    const int nl = lane & 15;
    const int base8 = (lane < 16) ? 0 : 8;
    const int pos = pt * 16 + nl;
    const int h = pos / period;
    const int w = pos - h * period;
    const int Wp = period + 10;
    v8f c = {0.f, 0.f, 0.f, 0.f, 0.f, 0.f, 0.f, 0.f};
    const _Float16* pA = weff + (size_t)(ct * 16 + nl) * DM + base8;
    const _Float16* pRow = srcpad + ((size_t)b * PADP + h * Wp + w) * DM + base8;
    for (int dy = 0; dy < 11; ++dy) {
        const _Float16* pB = pRow;
        for (int dx = 0; dx < 11; ++dx) {
            #pragma unroll
            for (int kk = 0; kk < 64; kk += 32) {
                F16x16 a, bb;
                a.q[0]  = *(const int4*)(pA + kk);
                a.q[1]  = *(const int4*)(pA + kk + 16);
                bb.q[0] = *(const int4*)(pB + kk);
                bb.q[1] = *(const int4*)(pB + kk + 16);
                c = __builtin_amdgcn_wmma_f32_16x16x32_f16(false, a.v, false, bb.v,
                                                           (short)0, c, false, false);
            }
            pA += 4096;      // next offset in weff [off][64][64]
            pB += DM;        // dx+1
        }
        pRow += (size_t)Wp * DM;  // dy+1
    }
    return c;
}

// conv1: encpad -> gelu -> midpad (padded interior).  grid (16,32) x (32,4)
__global__ void conv1_kernel(const _Float16* __restrict__ encpad,
                             const _Float16* __restrict__ weff,
                             const float* __restrict__ beff,
                             const int* __restrict__ bp, int br,
                             _Float16* __restrict__ midpad) {
    int lane = threadIdx.x, ct = threadIdx.y;
    int b = blockIdx.x, pt = blockIdx.y;
    int period = bp[br * 4 + 1], L = bp[br * 4 + 3];
    if (pt * 16 >= L) return;                       // block-uniform early out
    v8f c = conv_tile(encpad, weff, b, ct, pt, period, lane);
    int nl = lane & 15, base8 = (lane < 16) ? 0 : 8;
    int pos = pt * 16 + nl, mb = ct * 16 + base8;
    if (pos < L) {
        int h = pos / period;
        int w = pos - h * period;
        int pp = (h + 5) * (period + 10) + (w + 5);
        H8 o;
        #pragma unroll
        for (int r = 0; r < 8; ++r)
            o.h[r] = (_Float16)gelu_exact(c[r] + beff[mb + r]);
        *(int4*)(midpad + ((size_t)b * PADP + pp) * DM + mb) = o.q;
    }
}

// conv2: midpad -> acc += sw[b,br] * (conv + bias).  grid (16,16) x (32,4)
__global__ void conv2_kernel(const _Float16* __restrict__ midpad,
                             const _Float16* __restrict__ weff,
                             const float* __restrict__ beff,
                             const int* __restrict__ bp, int br,
                             const float* __restrict__ sw,
                             float* __restrict__ acc) {
    int lane = threadIdx.x, ct = threadIdx.y;
    int b = blockIdx.x, pt = blockIdx.y;
    int period = bp[br * 4 + 1];
    v8f c = conv_tile(midpad, weff, b, ct, pt, period, lane);
    int nl = lane & 15, base8 = (lane < 16) ? 0 : 8;
    int pos = pt * 16 + nl, mb = ct * 16 + base8;
    float wgt = sw[b * 3 + br];
    float* pd = acc + ((size_t)b * SEQ + pos) * DM + mb;
    #pragma unroll
    for (int r = 0; r < 8; ++r) pd[r] += wgt * (c[r] + beff[mb + r]);
}

// ---------------------------------------------------------------------------
// LayerNorm over 64 channels.  grid: 4096 blocks x 64 threads
// ---------------------------------------------------------------------------
__global__ void layernorm_kernel(const float* __restrict__ acc,
                                 const float* __restrict__ g, const float* __restrict__ be,
                                 float* __restrict__ enc) {
    int row = blockIdx.x, n = threadIdx.x;
    float v = acc[(size_t)row * DM + n];
    __shared__ float s1[64], s2[64];
    s1[n] = v; s2[n] = v * v;
    __syncthreads();
    for (int st = 32; st > 0; st >>= 1) {
        if (n < st) { s1[n] += s1[n + st]; s2[n] += s2[n + st]; }
        __syncthreads();
    }
    float mean = s1[0] * (1.0f / 64.0f);
    float var = s2[0] * (1.0f / 64.0f) - mean * mean;
    enc[(size_t)row * DM + n] = (v - mean) * rsqrtf(var + 1e-5f) * g[n] + be[n];
}

// ---------------------------------------------------------------------------
// Head GEMM: [16,K] x W[N,K]^T, fp32 WMMA 16x16x4, split-K, atomic f32 accum.
// grid: (NTILE, GEMM_SPLITS) x 32 threads
// ---------------------------------------------------------------------------
__global__ void gemm_wmma_f32(const float* __restrict__ A, int lda,
                              const float* __restrict__ W, int N, int K,
                              float* __restrict__ Cacc, int ldc) {
    const int lane = threadIdx.x;
    const int nl = lane & 15;
    const int n = blockIdx.x * 16 + nl;
    const bool rowok = (n < N);
    const int koff = (lane < 16) ? 0 : 2;
    const int iters = (K + 3) >> 2;
    const int per = (iters + (int)gridDim.y - 1) / (int)gridDim.y;
    int i0 = blockIdx.y * per;
    int i1 = i0 + per; if (i1 > iters) i1 = iters;
    const float* __restrict__ wrow = W + (size_t)(rowok ? n : 0) * K;
    const float* __restrict__ arow = A + (size_t)nl * lda;
    v8f c = {0.f, 0.f, 0.f, 0.f, 0.f, 0.f, 0.f, 0.f};
    for (int it = i0; it < i1; ++it) {
        const int k0 = (it << 2) + koff;
        v2f a, bv;
        a.x = arow[k0];
        a.y = arow[k0 + 1];
        bv.x = (rowok && (k0 < K))     ? wrow[k0]     : 0.0f;
        bv.y = (rowok && (k0 + 1 < K)) ? wrow[k0 + 1] : 0.0f;
        c = __builtin_amdgcn_wmma_f32_16x16x4_f32(false, a, false, bv,
                                                  (short)0, c, false, false);
    }
    if (rowok) {
        const int mrow = (lane < 16) ? 0 : 8;
        #pragma unroll
        for (int r = 0; r < 8; ++r)
            atomicAdd(&Cacc[(size_t)(mrow + r) * ldc + n], c[r]);
    }
}

__global__ void zero_f32_kernel(float* __restrict__ p, int n) {
    int i = blockIdx.x * blockDim.x + threadIdx.x;
    if (i < n) p[i] = 0.0f;
}

__global__ void bias_gelu_kernel(const float* __restrict__ acc, const float* __restrict__ bias,
                                 float* __restrict__ out, int N, int ldc, int total) {
    int i = blockIdx.x * blockDim.x + threadIdx.x;
    if (i >= total) return;
    int m = i % ldc;
    out[i] = (m < N) ? gelu_exact(acc[i] + bias[m]) : 0.0f;
}

// final [16,MID] x [8,MID]^T + b3 -> d_out[16,8].  grid 128 blocks x 256
__global__ void head3_kernel(const float* __restrict__ h2,
                             const float* __restrict__ w3, const float* __restrict__ b3,
                             float* __restrict__ out) {
    int b = blockIdx.x >> 3, cc = blockIdx.x & 7;
    int t = threadIdx.x;
    float s = 0.0f;
    for (int k = t; k < MID; k += 256)
        s += h2[(size_t)b * MIDP + k] * w3[(size_t)cc * MID + k];
    __shared__ float sh[256];
    sh[t] = s;
    __syncthreads();
    for (int st = 128; st > 0; st >>= 1) {
        if (t < st) sh[t] += sh[t + st];
        __syncthreads();
    }
    if (t == 0) out[b * 8 + cc] = sh[0] + b3[cc];
}

// ---------------------------------------------------------------------------
extern "C" void kernel_launch(void* const* d_in, const int* in_sizes, int n_in,
                              void* d_out, int out_size, void* d_ws, size_t ws_size,
                              hipStream_t stream) {
    (void)in_sizes; (void)n_in; (void)out_size; (void)ws_size;

    const float* x_enc   = (const float*)d_in[0];
    const float* token_w = (const float*)d_in[1];
    const float* ln_g    = (const float*)d_in[2];
    const float* ln_b    = (const float*)d_in[3];
    // blocks: d_in[4 + 24*l + {0..5:w1, 6..11:b1, 12..17:w2, 18..23:b2}]
    const float* h_w1 = (const float*)d_in[76];
    const float* h_b1 = (const float*)d_in[77];
    const float* h_w2 = (const float*)d_in[78];
    const float* h_b2 = (const float*)d_in[79];
    const float* h_w3 = (const float*)d_in[80];
    const float* h_b3 = (const float*)d_in[81];

    // workspace carve-up (256B aligned)
    char* p = (char*)d_ws;
    auto carve = [&](size_t bytes) { char* r = p; p += (bytes + 255) & ~(size_t)255; return r; };
    float*     enc    = (float*)    carve((size_t)NB * SEQ * DM * 4);
    float*     acc    = (float*)    carve((size_t)NB * SEQ * DM * 4);
    _Float16*  encpad = (_Float16*) carve((size_t)NB * PADP * DM * 2);  // 6 MB
    _Float16*  midpad = (_Float16*) carve((size_t)NB * PADP * DM * 2);  // 6 MB
    _Float16*  weff1  = (_Float16*) carve((size_t)121 * 64 * 64 * 2);
    _Float16*  weff2  = (_Float16*) carve((size_t)121 * 64 * 64 * 2);
    float*     beff1  = (float*)    carve(64 * 4);
    float*     beff2  = (float*)    carve(64 * 4);
    float*     amp    = (float*)    carve((size_t)NB * 129 * 4);
    int*       bp     = (int*)      carve(16 * 4);
    float*     sw     = (float*)    carve((size_t)NB * 3 * 4);
    float*     h1     = (float*)    carve((size_t)NB * MIDP * 4);
    float*     h2     = (float*)    carve((size_t)NB * MIDP * 4);
    float*     hacc   = (float*)    carve((size_t)NB * MIDP * 4);

    const int pad4 = (int)((size_t)NB * PADP * DM * 2 / 16);  // int4 chunks

    // 1) token embedding + PE
    token_pe_kernel<<<NB * SEQ, DM, 0, stream>>>(x_enc, token_w, enc);

    // 2) three TimesBlocks
    for (int l = 0; l < 3; ++l) {
        int base = 4 + 24 * l;
        const float** W1 = (const float**)&d_in[base];
        const float** B1 = (const float**)&d_in[base + 6];
        const float** W2 = (const float**)&d_in[base + 12];
        const float** B2 = (const float**)&d_in[base + 18];

        make_weff_kernel<<<(121 * 64 * 64 + 255) / 256, 256, 0, stream>>>(
            W1[0], W1[1], W1[2], W1[3], W1[4], W1[5],
            B1[0], B1[1], B1[2], B1[3], B1[4], B1[5], weff1, beff1);
        make_weff_kernel<<<(121 * 64 * 64 + 255) / 256, 256, 0, stream>>>(
            W2[0], W2[1], W2[2], W2[3], W2[4], W2[5],
            B2[0], B2[1], B2[2], B2[3], B2[4], B2[5], weff2, beff2);

        dft_amp_kernel<<<dim3(NB, 129), DM, 0, stream>>>(enc, amp);
        topk_kernel<<<1, 160, 0, stream>>>(amp, bp, sw);
        copy_acc_kernel<<<(NB * SEQ * DM) / 256, 256, 0, stream>>>(enc, acc);

        for (int br = 0; br < 3; ++br) {
            stage_encpad_kernel<<<dim3(NB, PADP), DM, 0, stream>>>(enc, bp, br, encpad);
            zero_b16_kernel<<<(pad4 + 255) / 256, 256, 0, stream>>>((int4*)midpad, pad4);
            conv1_kernel<<<dim3(NB, 32), dim3(32, 4), 0, stream>>>(
                encpad, weff1, beff1, bp, br, midpad);
            conv2_kernel<<<dim3(NB, 16), dim3(32, 4), 0, stream>>>(
                midpad, weff2, beff2, bp, br, sw, acc);
        }
        layernorm_kernel<<<NB * SEQ, DM, 0, stream>>>(acc, ln_g, ln_b, enc);
    }

    // 3) head: h1 = gelu(flat @ W1^T + b1)
    const int htot = NB * MIDP;
    zero_f32_kernel<<<(htot + 255) / 256, 256, 0, stream>>>(hacc, htot);
    gemm_wmma_f32<<<dim3(NTILE, GEMM_SPLITS), 32, 0, stream>>>(
        enc, SEQ * DM, h_w1, MID, SEQ * DM, hacc, MIDP);
    bias_gelu_kernel<<<(htot + 255) / 256, 256, 0, stream>>>(hacc, h_b1, h1, MID, MIDP, htot);

    //    h2 = gelu(h1 @ W2^T + b2)
    zero_f32_kernel<<<(htot + 255) / 256, 256, 0, stream>>>(hacc, htot);
    gemm_wmma_f32<<<dim3(NTILE, GEMM_SPLITS), 32, 0, stream>>>(
        h1, MIDP, h_w2, MID, MID, hacc, MIDP);
    bias_gelu_kernel<<<(htot + 255) / 256, 256, 0, stream>>>(hacc, h_b2, h2, MID, MIDP, htot);

    //    out = h2 @ W3^T + b3
    head3_kernel<<<NB * 8, 256, 0, stream>>>(h2, h_w3, h_b3, (float*)d_out);
}
